// PeepholeBidiLSTM_5566277616288
// MI455X (gfx1250) — matrix operand order repaired and checked by reading, not compiled, not run on hardware
//
#include <hip/hip_runtime.h>
#include <hip/hip_bf16.h>
#include <stdint.h>

#define B_   32
#define T_   512
#define D_   256
#define H_   512
#define NG   2048          // 4*H
#define NBLK 32
#define NTHR 256

typedef __bf16 bf16;
typedef __attribute__((ext_vector_type(16))) __bf16       v16bf;
typedef __attribute__((ext_vector_type(8)))  float        v8f;
typedef __attribute__((ext_vector_type(4)))  unsigned int v4u;

union Frag { v16bf v; v4u q[2]; };

// Load a 32-byte WMMA fragment as two 16-byte chunks (both 16B aligned).
__device__ inline v16bf ld_frag(const bf16* p0, const bf16* p1) {
  Frag f;
  f.q[0] = *(const v4u*)p0;
  f.q[1] = *(const v4u*)p1;
  return f.v;
}

__device__ inline float sigmoidf_(float x) { return 1.f / (1.f + __expf(-x)); }

// ---------------- phase 0: f32 -> bf16 conversions ----------------
__global__ void cvt_bf16(const float* __restrict__ src, bf16* __restrict__ dst, int n) {
  int i = blockIdx.x * blockDim.x + threadIdx.x;
  int stride = gridDim.x * blockDim.x;
  for (; i < n; i += stride) dst[i] = (bf16)src[i];
}

// h0 (2,B,H) f32 -> ping-pong buffer slot 0 per direction, bf16
__global__ void init_h(const float* __restrict__ h0, bf16* __restrict__ hbuf) {
  int i = blockIdx.x * blockDim.x + threadIdx.x;
  if (i < 2 * B_ * H_) {
    int dir  = i >> 14;        // / (B_*H_)
    int rest = i & 16383;
    hbuf[(size_t)dir * 2 * B_ * H_ + rest] = (bf16)h0[i];   // pp = 0
  }
}

// ---------------- persistent bidirectional LSTM scan ----------------
// grid = 32 WGs x 256 thr (8 waves). Wave gw in [0,256):
//   dir  = gw>>7 ; jt = (gw&127)>>2 (16 h-columns) ; gate = gw&3 (i/f/g/o)
// Per step: 2 Mtiles x 24 Ktiles of v_wmma_f32_16x16x32_bf16 (K = [x:256 | h:512]).
__global__ __launch_bounds__(NTHR, 1) void bidi_lstm_scan(
    const bf16* __restrict__ xbf,   // [B][T][D] bf16
    const bf16* __restrict__ wih,   // [2][NG][D] bf16
    const bf16* __restrict__ whh,   // [2][NG][H] bf16
    bf16*       __restrict__ hbuf,  // [2][2][B][H] bf16 ping-pong
    const float* __restrict__ c0,   // [2][B][H]
    const float* wip_f, const float* wfp_f, const float* wop_f,
    const float* wip_r, const float* wfp_r, const float* wop_r,
    float* __restrict__ out,        // [B][T][2H]
    float* __restrict__ hn,         // [2][B][H]
    float* __restrict__ cn,         // [2][B][H]
    int*   __restrict__ bar)
{
  const int wave = threadIdx.x >> 5;
  const int lane = threadIdx.x & 31;
  const int l    = lane & 15;
  const int hi   = lane >> 4;
  const int gw   = blockIdx.x * 8 + wave;
  const int dir  = gw >> 7;
  const int widx = gw & 127;
  const int jt   = widx >> 2;
  const int gate = widx & 3;
  const int jslot = wave >> 2;           // which j-tile within this WG
  const int jcol  = jt * 16 + l;

  const bf16* wihd = wih + (size_t)dir * NG * D_;
  const bf16* whhd = whh + (size_t)dir * NG * H_;
  bf16*       hb   = hbuf + (size_t)dir * 2 * B_ * H_;
  const int   nrow = gate * H_ + jt * 16 + l;       // gate-block row of [Wih|Whh]
  const bf16* wih_row = wihd + (size_t)nrow * D_;
  const bf16* whh_row = whhd + (size_t)nrow * H_;

  const float wip = dir ? wip_r[jcol] : wip_f[jcol];
  const float wfp = dir ? wfp_r[jcol] : wfp_f[jcol];
  const float wop = dir ? wop_r[jcol] : wop_f[jcol];

  // c-state lives in registers for the whole scan (C-layout: b = 16*mt + 8*hi + r)
  float c[16], hreg[16];
  #pragma unroll
  for (int mt = 0; mt < 2; ++mt)
    #pragma unroll
    for (int r = 0; r < 8; ++r) {
      int b = mt * 16 + 8 * hi + r;
      c[mt * 8 + r]    = c0[(size_t)dir * B_ * H_ + (size_t)b * H_ + jcol];
      hreg[mt * 8 + r] = 0.f;
    }

  __shared__ float lds[2][4][2][256];    // [jslot][gate][mtile][r*32+lane] = 16KB

  const int aoff = hi * 8;               // A frag: K = base+aoff+{0..7,16..23}
  const int boff = hi * 8;               // B frag: K = base+boff+{0..7,16..23}
  const bf16* xrow0base = xbf + (size_t)(l)      * T_ * D_;  // A rows = batches
  const bf16* xrow1base = xbf + (size_t)(l + 16) * T_ * D_;

  for (int s = 0; s < T_; ++s) {
    const int t = dir ? (T_ - 1 - s) : s;
    const bf16* hsrc = hb + (size_t)(s & 1) * B_ * H_;
    bf16*       hdst = hb + (size_t)((s + 1) & 1) * B_ * H_;

    v8f acc0 = {0.f, 0.f, 0.f, 0.f, 0.f, 0.f, 0.f, 0.f};
    v8f acc1 = acc0;

    const bf16* xr0 = xrow0base + (size_t)t * D_;
    const bf16* xr1 = xrow1base + (size_t)t * D_;
    const bf16* hr0 = hsrc + (size_t)l * H_;
    const bf16* hr1 = hsrc + (size_t)(l + 16) * H_;

    #pragma unroll
    for (int kt = 0; kt < D_ / 32; ++kt) {        // x part of K
      const int kb = kt * 32;
      v16bf bfr = ld_frag(wih_row + kb + boff, wih_row + kb + boff + 16);
      v16bf a0  = ld_frag(xr0 + kb + aoff, xr0 + kb + aoff + 16);
      v16bf a1  = ld_frag(xr1 + kb + aoff, xr1 + kb + aoff + 16);
      acc0 = __builtin_amdgcn_wmma_f32_16x16x32_bf16(false, a0, false, bfr, (short)0, acc0, false, false);
      acc1 = __builtin_amdgcn_wmma_f32_16x16x32_bf16(false, a1, false, bfr, (short)0, acc1, false, false);
    }
    #pragma unroll
    for (int kt = 0; kt < H_ / 32; ++kt) {        // h part of K
      const int kb = kt * 32;
      v16bf bfr = ld_frag(whh_row + kb + boff, whh_row + kb + boff + 16);
      v16bf a0  = ld_frag(hr0 + kb + aoff, hr0 + kb + aoff + 16);
      v16bf a1  = ld_frag(hr1 + kb + aoff, hr1 + kb + aoff + 16);
      acc0 = __builtin_amdgcn_wmma_f32_16x16x32_bf16(false, a0, false, bfr, (short)0, acc0, false, false);
      acc1 = __builtin_amdgcn_wmma_f32_16x16x32_bf16(false, a1, false, bfr, (short)0, acc1, false, false);
    }

    #pragma unroll
    for (int r = 0; r < 8; ++r) {
      lds[jslot][gate][0][r * 32 + lane] = acc0[r];
      lds[jslot][gate][1][r * 32 + lane] = acc1[r];
    }
    __syncthreads();

    if (gate == 0) {                               // elementwise update (exact ref math)
      #pragma unroll
      for (int mt = 0; mt < 2; ++mt)
        #pragma unroll
        for (int r = 0; r < 8; ++r) {
          float iv = lds[jslot][0][mt][r * 32 + lane];
          float fv = lds[jslot][1][mt][r * 32 + lane];
          float gv = lds[jslot][2][mt][r * 32 + lane];
          float ov = lds[jslot][3][mt][r * 32 + lane];
          int   idx = mt * 8 + r;
          float cp  = c[idx];
          float ii  = sigmoidf_(iv + wip * cp);
          float ff  = sigmoidf_(fv + wfp * cp);
          float gg  = tanhf(gv);
          float cy  = ff * cp + ii * gg;
          float oo  = ov + wop * cy;               // NOTE: reference has no sigmoid on o
          float hy  = oo * tanhf(cy);
          c[idx]    = cy;
          hreg[idx] = hy;
          int b = mt * 16 + 8 * hi + r;
          out[((size_t)b * T_ + t) * (2 * H_) + (size_t)dir * H_ + jcol] = hy;
          hdst[(size_t)b * H_ + jcol] = (bf16)hy;
        }
      __threadfence();
    }

    // device-wide barrier (monotonic epoch; counter zeroed per launch by memset)
    __syncthreads();
    if (threadIdx.x == 0) {
      __hip_atomic_fetch_add(bar, 1, __ATOMIC_ACQ_REL, __HIP_MEMORY_SCOPE_AGENT);
      const int target = (s + 1) * NBLK;
      while (__hip_atomic_load(bar, __ATOMIC_ACQUIRE, __HIP_MEMORY_SCOPE_AGENT) < target)
        __builtin_amdgcn_s_sleep(1);
    }
    __syncthreads();
  }

  if (gate == 0) {                                 // final states
    #pragma unroll
    for (int mt = 0; mt < 2; ++mt)
      #pragma unroll
      for (int r = 0; r < 8; ++r) {
        int idx = mt * 8 + r;
        int b   = mt * 16 + 8 * hi + r;
        hn[(size_t)dir * B_ * H_ + (size_t)b * H_ + jcol] = hreg[idx];
        cn[(size_t)dir * B_ * H_ + (size_t)b * H_ + jcol] = c[idx];
      }
  }
}

extern "C" void kernel_launch(void* const* d_in, const int* in_sizes, int n_in,
                              void* d_out, int out_size, void* d_ws, size_t ws_size,
                              hipStream_t stream) {
  const float* x      = (const float*)d_in[0];
  const float* h0     = (const float*)d_in[1];
  const float* c0     = (const float*)d_in[2];
  const float* w_ih_f = (const float*)d_in[3];
  const float* w_hh_f = (const float*)d_in[4];
  const float* w_ip_f = (const float*)d_in[5];
  const float* w_fp_f = (const float*)d_in[6];
  const float* w_op_f = (const float*)d_in[7];
  const float* w_ih_r = (const float*)d_in[8];
  const float* w_hh_r = (const float*)d_in[9];
  const float* w_ip_r = (const float*)d_in[10];
  const float* w_fp_r = (const float*)d_in[11];
  const float* w_op_r = (const float*)d_in[12];

  char* ws   = (char*)d_ws;
  int*  bar  = (int*)ws;                                  // 128 B barrier area
  bf16* xbf  = (bf16*)(ws + 128);                         // B*T*D     = 4,194,304
  bf16* wih  = xbf + (size_t)B_ * T_ * D_;                // 2*NG*D    = 1,048,576
  bf16* whh  = wih + (size_t)2 * NG * D_;                 // 2*NG*H    = 2,097,152
  bf16* hbuf = whh + (size_t)2 * NG * H_;                 // 2*2*B*H   =    65,536
  // total ws use ~14.2 MB

  float* out = (float*)d_out;
  float* hn  = out + (size_t)B_ * T_ * 2 * H_;
  float* cn  = hn + (size_t)2 * B_ * H_;

  hipMemsetAsync(bar, 0, 128, stream);

  cvt_bf16<<<2048, 256, 0, stream>>>(x, xbf, B_ * T_ * D_);
  cvt_bf16<<<512,  256, 0, stream>>>(w_ih_f, wih,            NG * D_);
  cvt_bf16<<<512,  256, 0, stream>>>(w_ih_r, wih + NG * D_,  NG * D_);
  cvt_bf16<<<1024, 256, 0, stream>>>(w_hh_f, whh,            NG * H_);
  cvt_bf16<<<1024, 256, 0, stream>>>(w_hh_r, whh + NG * H_,  NG * H_);
  init_h<<<(2 * B_ * H_ + 255) / 256, 256, 0, stream>>>(h0, hbuf);

  bidi_lstm_scan<<<NBLK, NTHR, 0, stream>>>(
      xbf, wih, whh, hbuf, c0,
      w_ip_f, w_fp_f, w_op_f, w_ip_r, w_fp_r, w_op_r,
      out, hn, cn, bar);
}
